// MemoryInterface_14955076125124
// MI455X (gfx1250) — matrix operand order repaired and checked by reading, not compiled, not run on hardware
//
#include <hip/hip_runtime.h>
#include <math.h>

typedef _Float16 v16h __attribute__((ext_vector_type(16)));
typedef _Float16 v8h  __attribute__((ext_vector_type(8)));
typedef float    v8f  __attribute__((ext_vector_type(8)));
typedef unsigned int u32x4 __attribute__((ext_vector_type(4)));
typedef int          i32x4 __attribute__((ext_vector_type(4)));
typedef int          i32x8 __attribute__((ext_vector_type(8)));

#define B_ROWS 64
#define DIM    128
#define CDIM   256
#define CHUNK  128     // keys per TDM tile
#define TOPK   8
#define GRID_SIM 128   // persistent blocks

// ---- LDS layout (bytes) ----
#define STG_FLOATS (CHUNK * DIM)                 // 16384 f32 per buffer
#define STG_BYTES  (2 * STG_FLOATS * 4)          // 131072 (double buffer)
#define KT_STRIDE  136                           // halves: 128 dims + 8 pad
#define QS_STRIDE  136
#define S_STRIDE   129                           // floats, bank-rotating
#define KT_BYTES   (DIM * KT_STRIDE * 2)         // 34816 (S overlays this)
#define QS_BYTES   (B_ROWS * QS_STRIDE * 2)      // 17408
#define SC_BYTES   (CHUNK * 4)                   // 512
#define SMEM_BYTES (STG_BYTES + KT_BYTES + QS_BYTES + SC_BYTES)  // 183808

#ifndef __has_builtin
#define __has_builtin(x) 0
#endif
#if __has_builtin(__builtin_amdgcn_tensor_load_to_lds)
#define HAVE_TDM 1
#else
#define HAVE_TDM 0
#endif

// -------- register-resident top-8 insertion ladder (descending) ----------
__device__ __forceinline__ void top8_insert(float (&tv)[TOPK], int (&ti)[TOPK],
                                            float v, int idx) {
    if (v <= tv[TOPK - 1]) return;
    float cv = v; int ci = idx;
#pragma unroll
    for (int p = TOPK - 1; p >= 0; --p) {
        if (cv > tv[p]) {
            float tf = tv[p]; tv[p] = cv; cv = tf;
            int   td = ti[p]; ti[p] = ci; ci = td;
        }
    }
}

#if HAVE_TDM
// Issue one TDM 2-D tile load: rows x 128 f32, global -> LDS (raw bytes).
// D# per CDNA5 ISA ch.8: group0 = {count/flags, lds_addr, global_addr, type},
// group1 = {data_size, tensor dims/strides, tile dims}. OOB rows read zero.
// Toolchain builtin is the 6-arg form: (u32x4, i32x8, i32x4, i32x4, i32x8, cpol).
__device__ __forceinline__ void tdm_issue_keys(const float* gsrc,
                                               unsigned lds_addr, int rows) {
    unsigned long long ga = (unsigned long long)(uintptr_t)gsrc;
    u32x4 g0;
    g0[0] = 1u;                                   // count=1, user mode
    g0[1] = lds_addr;                             // LDS byte address
    g0[2] = (unsigned)ga;                         // global_addr[31:0]
    g0[3] = ((unsigned)(ga >> 32) & 0x01FFFFFFu)  // global_addr[56:32]
            | 0x80000000u;                        // type=2 ("image")
    i32x8 g1;
    g1[0] = 0x00020000;                           // data_size=2 -> 4 bytes
    g1[1] = (int)((unsigned)DIM << 16);           // tensor_dim0 = 128 (lo16 in [31:16])
    g1[2] = (int)(((unsigned)rows & 0xFFFFu) << 16);            // tensor_dim1[15:0]
    g1[3] = (int)((((unsigned)rows >> 16) & 0xFFFFu)            // tensor_dim1[31:16]
                  | ((unsigned)DIM << 16));                     // tile_dim0 = 128
    g1[4] = CHUNK;                                // tile_dim1 = 128, tile_dim2 = 0
    g1[5] = DIM;                                  // tensor_dim0_stride = 128
    g1[6] = 0;
    g1[7] = 0;
    i32x4 z4 = {0, 0, 0, 0};
    i32x8 z8 = {0, 0, 0, 0, 0, 0, 0, 0};
    __builtin_amdgcn_tensor_load_to_lds(g0, g1, z4, z4, z8, 0);
}
#endif

// ------------------- Kernel 1: q = query @ W_q^T + b_q, L2-normalize, ->f16
__global__ __launch_bounds__(128)
void qproj_kernel(const float* __restrict__ query, const float* __restrict__ W_q,
                  const float* __restrict__ b_q, _Float16* __restrict__ qn) {
    __shared__ float red[DIM];
    const int b = blockIdx.x;
    const int d = threadIdx.x;          // 128 threads
    const float* qr = query + (size_t)b * CDIM;
    const float* wr = W_q   + (size_t)d * CDIM;
    float acc = b_q[d];
    for (int c = 0; c < CDIM; ++c) acc = fmaf(qr[c], wr[c], acc);
    red[d] = acc * acc;
    __syncthreads();
    for (int s = 64; s > 0; s >>= 1) {
        if (d < s) red[d] += red[d + s];
        __syncthreads();
    }
    const float nrm = sqrtf(red[0]);
    qn[(size_t)b * DIM + d] = (_Float16)(acc / fmaxf(nrm, 1e-12f));
}

// -- Kernel 2: persistent WMMA cosine-sim, TDM double-buffered key streaming
__global__ __launch_bounds__(256)
void sim_topk_kernel(const float* __restrict__ keys,
                     const float* __restrict__ importance,
                     const _Float16* __restrict__ qn,
                     float* __restrict__ pv, int* __restrict__ pi,
                     int nkeys, int nchunks) {
    extern __shared__ __align__(16) char smem[];
    float*    stg   = (float*)smem;                          // [2][CHUNK*DIM] f32
    _Float16* keysT = (_Float16*)(smem + STG_BYTES);         // [CHUNK][KT_STRIDE], key-major
    float*    S     = (float*)(smem + STG_BYTES);            // overlay on keysT
    _Float16* qs    = (_Float16*)(smem + STG_BYTES + KT_BYTES);
    float*    scale = (float*)(smem + STG_BYTES + KT_BYTES + QS_BYTES);

    const int tid  = threadIdx.x;
    const int wave = tid >> 5, lane = tid & 31;
    const int lhi  = lane >> 4, llo = lane & 15;
    const int mt   = wave & 3;            // M tile (rows 16*mt..)
    const int ntb  = (wave >> 2) * 4;     // N tile base (4 tiles per wave)

    // qn tile -> LDS once (dword copies)
    {
        const uint32_t* src = (const uint32_t*)qn;           // 64*64 dwords
        uint32_t* dst = (uint32_t*)qs;
        for (int i = tid; i < B_ROWS * (DIM / 2); i += 256) {
            int r = i >> 6, c = i & 63;
            dst[r * (QS_STRIDE / 2) + c] = src[i];
        }
    }

    // running per-row top-8 (threads 0..63, registers, persists across chunks)
    float tv[TOPK]; int ti[TOPK];
#pragma unroll
    for (int k = 0; k < TOPK; ++k) { tv[k] = -__builtin_inff(); ti[k] = 0; }

#if HAVE_TDM
    const unsigned lds_stg0 = (unsigned)(uintptr_t)(void*)&stg[0];
    const unsigned lds_stg1 = (unsigned)(uintptr_t)(void*)&stg[STG_FLOATS];
    if (wave == 0 && (int)blockIdx.x < nchunks) {
        const int c0 = blockIdx.x;
        tdm_issue_keys(keys + (size_t)c0 * CHUNK * DIM, lds_stg0,
                       nkeys - c0 * CHUNK);
    }
#endif
    __syncthreads();   // qs ready (TDM handoff is via tensorcnt below)

    int buf = 0;
    for (int chunk = blockIdx.x; chunk < nchunks; chunk += GRID_SIM) {
#if HAVE_TDM
        if (wave == 0) {
            __builtin_amdgcn_s_wait_tensorcnt(0);   // stg[buf] ready
            const int next = chunk + GRID_SIM;
            if (next < nchunks)
                tdm_issue_keys(keys + (size_t)next * CHUNK * DIM,
                               buf ? lds_stg0 : lds_stg1, nkeys - next * CHUNK);
        }
        __syncthreads();
#else
        // fallback: cooperative synchronous fill of stg[buf]
        for (int i = tid; i < CHUNK * DIM; i += 256) {
            const int g = chunk * CHUNK + (i >> 7);
            stg[buf * STG_FLOATS + i] =
                (g < nkeys) ? keys[(size_t)chunk * CHUNK * DIM + i] : 0.0f;
        }
        __syncthreads();
#endif
        const float* sb = stg + buf * STG_FLOATS;

        // ---- convert f32 staging -> f16 keysT, same [key][dim] order (overwrites old S)
        for (int i = tid; i < CHUNK * DIM; i += 256) {
            const int kk = i >> 7, dd = i & (DIM - 1);
            keysT[kk * KT_STRIDE + dd] = (_Float16)sb[i];
        }
        // ---- per-key scale = importance / max(||k||, eps)
        if (tid < CHUNK) {
            float ss = 0.0f;
            for (int dd = 0; dd < DIM; ++dd) {
                const int d2 = (dd + tid) & (DIM - 1);   // rotate start (banks)
                const float v = sb[tid * DIM + d2];
                ss = fmaf(v, v, ss);
            }
            const int g = chunk * CHUNK + tid;
            const float imp = (g < nkeys) ? importance[g] : 0.0f;
            scale[tid] = imp / fmaxf(sqrtf(ss), 1e-12f);
        }
        __syncthreads();

        // ---- WMMA: S[64,128] = qn(f16) x keysT(f16)
        v8f zero = {0.f, 0.f, 0.f, 0.f, 0.f, 0.f, 0.f, 0.f};
        v8f acc[4];
#pragma unroll
        for (int j = 0; j < 4; ++j) acc[j] = zero;

#pragma unroll
        for (int kb = 0; kb < 4; ++kb) {
            const int arow  = mt * 16 + llo;
            const int abase = kb * 32 + lhi * 8;
            v8h alo = *(const v8h*)&qs[arow * QS_STRIDE + abase];
            v8h ahi = *(const v8h*)&qs[arow * QS_STRIDE + abase + 16];
            v16h a = __builtin_shufflevector(alo, ahi,
                     0, 1, 2, 3, 4, 5, 6, 7, 8, 9, 10, 11, 12, 13, 14, 15);
#pragma unroll
            for (int j = 0; j < 4; ++j) {
                const int brow = (ntb + j) * 16 + llo; // N = key row, K along elements
                const int bcol = kb * 32 + lhi * 8;
                v8h blo = *(const v8h*)&keysT[brow * KT_STRIDE + bcol];
                v8h bhi = *(const v8h*)&keysT[brow * KT_STRIDE + bcol + 16];
                v16h bm = __builtin_shufflevector(blo, bhi,
                          0, 1, 2, 3, 4, 5, 6, 7, 8, 9, 10, 11, 12, 13, 14, 15);
                acc[j] = __builtin_amdgcn_wmma_f32_16x16x32_f16(
                             false, a, false, bm, (short)0, acc[j], false, false);
            }
        }
        __syncthreads();   // WMMA LDS reads done before overlaying S on keysT

        // ---- scale + write S tile
#pragma unroll
        for (int j = 0; j < 4; ++j) {
            const int col = (ntb + j) * 16 + llo;
            const float sc = scale[col];
#pragma unroll
            for (int r = 0; r < 8; ++r) {
                const int row = mt * 16 + r + 8 * lhi;
                S[row * S_STRIDE + col] = acc[j][r] * sc;
            }
        }
        __syncthreads();

        // ---- accumulate running top-8 over this chunk
        if (tid < B_ROWS) {
            for (int c = 0; c < CHUNK; ++c) {
                const int g = chunk * CHUNK + c;
                if (g >= nkeys) break;
                top8_insert(tv, ti, S[tid * S_STRIDE + c], g);
            }
        }
        __syncthreads();   // S / staging buffer safe to reuse
        buf ^= 1;
    }

    if (tid < B_ROWS) {
        const int base = (blockIdx.x * B_ROWS + tid) * TOPK;
#pragma unroll
        for (int k = 0; k < TOPK; ++k) { pv[base + k] = tv[k]; pi[base + k] = ti[k]; }
    }
}

// --------- Kernel 3: merge partial top-8 lists, softmax attention, combine
__global__ __launch_bounds__(256)
void merge_attn_kernel(const float* __restrict__ pv, const int* __restrict__ pi,
                       int nlists,
                       const float* __restrict__ values,
                       const float* __restrict__ w_attn,
                       const float* __restrict__ b_attn,
                       float* __restrict__ mo) {
    __shared__ float sc[B_ROWS][TOPK];
    __shared__ int   si[B_ROWS][TOPK];
    const int tid = threadIdx.x;

    if (tid < B_ROWS) {
        float tv[TOPK]; int ti[TOPK];
#pragma unroll
        for (int k = 0; k < TOPK; ++k) { tv[k] = -__builtin_inff(); ti[k] = 0; }
        for (int blk = 0; blk < nlists; ++blk) {
            const int base = (blk * B_ROWS + tid) * TOPK;
#pragma unroll
            for (int k = 0; k < TOPK; ++k) {
                const float v = pv[base + k];
                if (v <= tv[TOPK - 1]) break;   // lists sorted descending
                top8_insert(tv, ti, v, pi[base + k]);
            }
        }
        float lg[TOPK];
        float mx = -__builtin_inff();
        const float ba = b_attn[0];
#pragma unroll
        for (int k = 0; k < TOPK; ++k) {
            const float* vr = values + (size_t)ti[k] * DIM;
            float a = ba;
            for (int d = 0; d < DIM; ++d) a = fmaf(vr[d], w_attn[d], a);
            lg[k] = a; mx = fmaxf(mx, a);
        }
        float den = 0.0f;
#pragma unroll
        for (int k = 0; k < TOPK; ++k) { lg[k] = expf(lg[k] - mx); den += lg[k]; }
        const float inv = 1.0f / den;
#pragma unroll
        for (int k = 0; k < TOPK; ++k) { sc[tid][k] = lg[k] * inv; si[tid][k] = ti[k]; }
    }
    __syncthreads();

    for (int e = tid; e < B_ROWS * DIM; e += 256) {
        const int b = e >> 7, d = e & (DIM - 1);
        float s = 0.0f;
#pragma unroll
        for (int k = 0; k < TOPK; ++k)
            s = fmaf(sc[b][k], values[(size_t)si[b][k] * DIM + d], s);
        mo[e] = s;
    }
}

// ---------------- Kernel 4: out = mo @ W_comb^T + b_comb  (tiny GEMM)
__global__ __launch_bounds__(256)
void out_kernel(const float* __restrict__ mo, const float* __restrict__ W_comb,
                const float* __restrict__ b_comb, float* __restrict__ out) {
    const int e = blockIdx.x * blockDim.x + threadIdx.x;
    if (e >= B_ROWS * DIM) return;
    const int b = e >> 7, d = e & (DIM - 1);
    const float* m = mo + (size_t)b * DIM;
    const float* w = W_comb + (size_t)d * DIM;
    float acc = b_comb[d];
    for (int j = 0; j < DIM; ++j) acc = fmaf(m[j], w[j], acc);
    out[e] = acc;
}

extern "C" void kernel_launch(void* const* d_in, const int* in_sizes, int n_in,
                              void* d_out, int out_size, void* d_ws, size_t ws_size,
                              hipStream_t stream) {
    const float* query      = (const float*)d_in[0];
    const float* keys       = (const float*)d_in[1];
    const float* values     = (const float*)d_in[2];
    const float* importance = (const float*)d_in[3];
    const float* W_q        = (const float*)d_in[4];
    const float* b_q        = (const float*)d_in[5];
    const float* w_attn     = (const float*)d_in[6];
    const float* b_attn     = (const float*)d_in[7];
    const float* W_comb     = (const float*)d_in[8];
    const float* b_comb     = (const float*)d_in[9];

    const int nkeys   = in_sizes[1] / DIM;
    const int nchunks = (nkeys + CHUNK - 1) / CHUNK;

    // workspace carve-out
    char* ws = (char*)d_ws;
    _Float16* qn = (_Float16*)ws;                               // 16 KB
    const size_t qn_bytes = (size_t)B_ROWS * DIM * sizeof(_Float16);
    float* pv = (float*)(ws + qn_bytes);
    const size_t part_bytes = (size_t)GRID_SIM * B_ROWS * TOPK * sizeof(float);
    int*   pi = (int*)(ws + qn_bytes + part_bytes);
    float* mo = (float*)(ws + qn_bytes + 2 * part_bytes);

    qproj_kernel<<<B_ROWS, DIM, 0, stream>>>(query, W_q, b_q, qn);
    sim_topk_kernel<<<GRID_SIM, 256, SMEM_BYTES, stream>>>(keys, importance, qn,
                                                           pv, pi, nkeys, nchunks);
    merge_attn_kernel<<<1, 256, 0, stream>>>(pv, pi, GRID_SIM, values,
                                             w_attn, b_attn, mo);
    out_kernel<<<(B_ROWS * DIM + 255) / 256, 256, 0, stream>>>(mo, W_comb,
                                                               b_comb,
                                                               (float*)d_out);
}
